// CrossNATBlock2D_84224308675135
// MI455X (gfx1250) — matrix operand order repaired and checked by reading, not compiled, not run on hardware
//
#include <hip/hip_runtime.h>

typedef unsigned int   u32;
typedef unsigned short u16;

typedef __bf16 bf16x16 __attribute__((ext_vector_type(16)));
typedef float  f32x8   __attribute__((ext_vector_type(8)));

union Frag {
    uint4   q[2];   // 32 bytes = 16 bf16
    bf16x16 v;
};

// ---------- bf16 helpers (round-to-nearest-even pack; shift unpack) ----------
__device__ __forceinline__ u16 f2bf(float f) {
    u32 u = __builtin_bit_cast(u32, f);
    u32 r = (u + 0x7FFFu + ((u >> 16) & 1u)) >> 16;
    return (u16)r;
}
__device__ __forceinline__ float bf_lo(u32 w) { return __builtin_bit_cast(float, w << 16); }
__device__ __forceinline__ float bf_hi(u32 w) { return __builtin_bit_cast(float, w & 0xFFFF0000u); }

// ---------- async global -> LDS copy (GLOBAL_LOAD_ASYNC_TO_LDS_B128) ----------
__device__ __forceinline__ void async_g2l_b128(u32 lds_byte_addr, const void* gaddr) {
    asm volatile("global_load_async_to_lds_b128 %0, %1, off"
                 :: "v"(lds_byte_addr), "v"(gaddr)
                 : "memory");
}
__device__ __forceinline__ void wait_asynccnt0() {
    asm volatile("s_wait_asynccnt 0x0" ::: "memory");
}

// ---------- f32 -> bf16 convert ----------
__global__ void cvt_f32_bf16(const float* __restrict__ src, u16* __restrict__ dst, int n) {
    int i = blockIdx.x * 256 + threadIdx.x;
    if (i < n) dst[i] = f2bf(src[i]);
}

// ---------- LayerNorm over C=128, one wave32 per pixel, bf16 output ----------
__global__ __launch_bounds__(256) void ln_bf16(const float* __restrict__ x,
                                               const float* __restrict__ w,
                                               const float* __restrict__ b,
                                               u16* __restrict__ out) {
    int pix  = blockIdx.x * 8 + (threadIdx.x >> 5);
    int lane = threadIdx.x & 31;
    const float* xp = x + (size_t)pix * 128;
    float v[4];
    float s = 0.f, s2 = 0.f;
    #pragma unroll
    for (int t = 0; t < 4; ++t) {
        float f = xp[lane + 32 * t];
        v[t] = f; s += f; s2 += f * f;
    }
    #pragma unroll
    for (int off = 16; off; off >>= 1) {
        s  += __shfl_xor(s,  off, 32);
        s2 += __shfl_xor(s2, off, 32);
    }
    float mu  = s * (1.f / 128.f);
    float var = s2 * (1.f / 128.f) - mu * mu;
    float inv = rsqrtf(var + 1e-5f);
    u16* op = out + (size_t)pix * 128;
    #pragma unroll
    for (int t = 0; t < 4; ++t) {
        int c = lane + 32 * t;
        op[c] = f2bf((v[t] - mu) * inv * w[c] + b[c]);
    }
}

// ---------- WMMA bf16 GEMM:  C[M,N] = A[M,K] @ W[N,K]^T + bias  (+epilogue) ----
// block = 256 threads = 8 waves; wave computes 32(M) x 64(N); block = 256M x 64N
// grid.x = M/256, grid.y = N/64.  KD: compile-time, multiple of 32.
// Register double-buffered: next K-step fragments prefetched before current WMMAs.
#define GM_STORE_BF16 0
#define GM_RESID_F32  1
#define GM_GELU_BF16  2

template <int KD, int MODE>
__global__ __launch_bounds__(256)
void gemm_wmma_bf16(const u16* __restrict__ A, const u16* __restrict__ W,
                    const float* __restrict__ bias, const float* __restrict__ resid,
                    u16* __restrict__ outb, float* __restrict__ outf, int N) {
    const int lane = threadIdx.x & 31;
    const int wave = threadIdx.x >> 5;
    const int m0   = blockIdx.x * 256 + wave * 32;
    const int n0   = blockIdx.y * 64;
    const int hl   = lane >> 4;   // lane-half selects K interleave
    const int r16  = lane & 15;

    f32x8 acc[2][4];
    #pragma unroll
    for (int mi = 0; mi < 2; ++mi)
        #pragma unroll
        for (int t = 0; t < 4; ++t)
            #pragma unroll
            for (int i = 0; i < 8; ++i) acc[mi][t][i] = 0.f;

    // A fragment: row = r16 (+16 for second M-tile), chunks at k+hl*8 and k+16+hl*8
    const u16* arow0 = A + (size_t)(m0 + r16) * KD + hl * 8;
    const u16* arow1 = arow0 + (size_t)16 * KD;
    // B fragment: col n = r16, chunks at k+hl*8 and k+16+hl*8 from weight row n
    const u16* wrow  = W + (size_t)(n0 + r16) * KD + hl * 8;

    Frag a[2], b[4];
    a[0].q[0] = *(const uint4*)(arow0);
    a[0].q[1] = *(const uint4*)(arow0 + 16);
    a[1].q[0] = *(const uint4*)(arow1);
    a[1].q[1] = *(const uint4*)(arow1 + 16);
    #pragma unroll
    for (int t = 0; t < 4; ++t) {
        const u16* p = wrow + (size_t)(t * 16) * KD;
        b[t].q[0] = *(const uint4*)(p);
        b[t].q[1] = *(const uint4*)(p + 16);
    }

    #pragma unroll
    for (int kb = 0; kb < KD; kb += 32) {
        Frag an[2], bn[4];
        if (kb + 32 < KD) {
            const int kn = kb + 32;
            an[0].q[0] = *(const uint4*)(arow0 + kn);
            an[0].q[1] = *(const uint4*)(arow0 + kn + 16);
            an[1].q[0] = *(const uint4*)(arow1 + kn);
            an[1].q[1] = *(const uint4*)(arow1 + kn + 16);
            #pragma unroll
            for (int t = 0; t < 4; ++t) {
                const u16* p = wrow + (size_t)(t * 16) * KD + kn;
                bn[t].q[0] = *(const uint4*)(p);
                bn[t].q[1] = *(const uint4*)(p + 16);
            }
        }
        #pragma unroll
        for (int mi = 0; mi < 2; ++mi)
            #pragma unroll
            for (int t = 0; t < 4; ++t)
                acc[mi][t] = __builtin_amdgcn_wmma_f32_16x16x32_bf16(
                    false, a[mi].v, false, b[t].v, (short)0, acc[mi][t], false, false);
        if (kb + 32 < KD) {
            a[0] = an[0]; a[1] = an[1];
            #pragma unroll
            for (int t = 0; t < 4; ++t) b[t] = bn[t];
        }
    }

    // C/D layout: reg r -> row (r + 8*hl), col = r16
    #pragma unroll
    for (int mi = 0; mi < 2; ++mi) {
        #pragma unroll
        for (int t = 0; t < 4; ++t) {
            int n = n0 + t * 16 + r16;
            float bs = bias[n];
            #pragma unroll
            for (int r = 0; r < 8; ++r) {
                int m = m0 + mi * 16 + r + 8 * hl;
                size_t o = (size_t)m * N + n;
                float val = acc[mi][t][r] + bs;
                if (MODE == GM_RESID_F32) {
                    outf[o] = val + resid[o];
                } else if (MODE == GM_GELU_BF16) {
                    outb[o] = f2bf(0.5f * val * (1.f + erff(val * 0.70710678118654752f)));
                } else {
                    outb[o] = f2bf(val);
                }
            }
        }
    }
}

// ---------- 7x7 neighborhood attention with boundary clamp ----------
// grid = (16, 16, 4 heads), block = 64 threads = one 8x8 pixel tile.
// qv: [16384, 256] bf16 (q cols 0..127, v cols 128..255), k: [16384,128] bf16.
__global__ __launch_bounds__(64)
void natten49(const u16* __restrict__ qv, const u16* __restrict__ kk,
              const float* __restrict__ rpb, u16* __restrict__ ao) {
    __shared__ u32 ks[14 * 14 * 16];   // halo keys,   head slice (32 bf16 = 16 dwords)
    __shared__ u32 vs[14 * 14 * 16];   // halo values
    const int tw = blockIdx.x * 8;
    const int th = blockIdx.y * 8;
    const int head = blockIdx.z;
    const int tid = threadIdx.x;

    const int h0 = max(th - 3, 0), h1 = min(th + 10, 127);
    const int w0 = max(tw - 3, 0), w1 = min(tw + 10, 127);
    const int nh = h1 - h0 + 1, nw = w1 - w0 + 1;

    // async halo copy: 16-byte units; per array nh*nw cells x 4 units
    const u32 ks_base = (u32)(uintptr_t)(&ks[0]);
    const u32 vs_base = (u32)(uintptr_t)(&vs[0]);
    const int total16 = nh * nw * 4;
    for (int i = tid; i < total16; i += 64) {
        int qq   = i & 3;               // 16-byte unit within 64-byte cell
        int cell = i >> 2;
        int rr = cell / nw, cc = cell - rr * nw;
        int gh = h0 + rr, gw = w0 + cc;
        u32 loff = (u32)(((rr * 14 + cc) * 16 + qq * 4) * 4);
        const void* gk = (const void*)(kk + ((size_t)(gh * 128 + gw) * 128 + head * 32 + qq * 8));
        const void* gv = (const void*)(qv + ((size_t)(gh * 128 + gw) * 256 + 128 + head * 32 + qq * 8));
        async_g2l_b128(ks_base + loff, gk);
        async_g2l_b128(vs_base + loff, gv);
    }
    wait_asynccnt0();
    __syncthreads();

    const int pi = tid >> 3, pj = tid & 7;
    const int i = th + pi, j = tw + pj;

    u32 qw[16];
    const u32* qp = (const u32*)(qv + ((size_t)(i * 128 + j) * 256 + head * 32));
    #pragma unroll
    for (int d = 0; d < 16; ++d) qw[d] = qp[d];

    const int sh = min(max(i - 3, 0), 121);
    const int sw = min(max(j - 3, 0), 121);
    const float scale = 0.17677669529663687f;   // 32^-0.5

    float logit[49];
    float mx = -1e30f;
    #pragma unroll
    for (int a = 0; a < 7; ++a) {
        #pragma unroll
        for (int c = 0; c < 7; ++c) {
            int gh = sh + a, gw = sw + c;
            const u32* kp = &ks[((gh - h0) * 14 + (gw - w0)) * 16];
            float acc = 0.f;
            #pragma unroll
            for (int d = 0; d < 16; ++d) {
                u32 kv = kp[d], q2 = qw[d];
                acc += bf_lo(q2) * bf_lo(kv) + bf_hi(q2) * bf_hi(kv);
            }
            float l = acc * scale + rpb[(head * 13 + (gh - i + 6)) * 13 + (gw - j + 6)];
            logit[a * 7 + c] = l;
            mx = fmaxf(mx, l);
        }
    }

    float accv[32];
    #pragma unroll
    for (int d = 0; d < 32; ++d) accv[d] = 0.f;
    float sum = 0.f;
    #pragma unroll
    for (int a = 0; a < 7; ++a) {
        #pragma unroll
        for (int c = 0; c < 7; ++c) {
            int gh = sh + a, gw = sw + c;
            float wgt = __expf(logit[a * 7 + c] - mx);
            sum += wgt;
            const u32* vp = &vs[((gh - h0) * 14 + (gw - w0)) * 16];
            #pragma unroll
            for (int d = 0; d < 16; ++d) {
                u32 vv = vp[d];
                accv[2 * d]     += wgt * bf_lo(vv);
                accv[2 * d + 1] += wgt * bf_hi(vv);
            }
        }
    }
    float inv = 1.f / sum;
    u32* op = (u32*)(ao + (size_t)(i * 128 + j) * 128 + head * 32);
    #pragma unroll
    for (int d = 0; d < 16; ++d) {
        op[d] = (u32)f2bf(accv[2 * d] * inv) | ((u32)f2bf(accv[2 * d + 1] * inv) << 16);
    }
}

// -------------------------------- launcher -----------------------------------
extern "C" void kernel_launch(void* const* d_in, const int* in_sizes, int n_in,
                              void* d_out, int out_size, void* d_ws, size_t ws_size,
                              hipStream_t stream) {
    (void)in_sizes; (void)n_in; (void)out_size; (void)ws_size;
    const float* x      = (const float*)d_in[0];
    const float* y      = (const float*)d_in[1];
    const float* qv_w   = (const float*)d_in[2];
    const float* qv_b   = (const float*)d_in[3];
    const float* k_w    = (const float*)d_in[4];
    const float* k_b    = (const float*)d_in[5];
    const float* rpb    = (const float*)d_in[6];
    const float* proj_w = (const float*)d_in[7];
    const float* proj_b = (const float*)d_in[8];
    const float* n1_w   = (const float*)d_in[9];
    const float* n1_b   = (const float*)d_in[10];
    const float* n2_w   = (const float*)d_in[11];
    const float* n2_b   = (const float*)d_in[12];
    const float* fc1_w  = (const float*)d_in[13];
    const float* fc1_b  = (const float*)d_in[14];
    const float* fc2_w  = (const float*)d_in[15];
    const float* fc2_b  = (const float*)d_in[16];

    const int NP = 128 * 128;   // 16384 pixels

    char* ws = (char*)d_ws;
    size_t off = 0;
    auto carve = [&](size_t bytes) -> char* {
        char* p = ws + off;
        off += (bytes + 255) & ~(size_t)255;
        return p;
    };
    u16*   y_bf   = (u16*)carve((size_t)NP * 128 * 2);
    u16*   xn_bf  = (u16*)carve((size_t)NP * 128 * 2);
    u16*   qvbf   = (u16*)carve((size_t)NP * 256 * 2);
    u16*   k_bf   = (u16*)carve((size_t)NP * 128 * 2);
    u16*   ao_bf  = (u16*)carve((size_t)NP * 128 * 2);
    float* x1     = (float*)carve((size_t)NP * 128 * 4);
    u16*   x1n_bf = (u16*)carve((size_t)NP * 128 * 2);
    u16*   h_bf   = (u16*)carve((size_t)NP * 512 * 2);
    u16*   qvw_bf = (u16*)carve(256 * 128 * 2);
    u16*   kw_bf  = (u16*)carve(128 * 128 * 2);
    u16*   pw_bf  = (u16*)carve(128 * 128 * 2);
    u16*   f1w_bf = (u16*)carve(512 * 128 * 2);
    u16*   f2w_bf = (u16*)carve(128 * 512 * 2);

    // ---- weight / activation converts to bf16 ----
    cvt_f32_bf16<<<(NP * 128 + 255) / 256, 256, 0, stream>>>(y, y_bf, NP * 128);
    cvt_f32_bf16<<<(256 * 128 + 255) / 256, 256, 0, stream>>>(qv_w, qvw_bf, 256 * 128);
    cvt_f32_bf16<<<(128 * 128 + 255) / 256, 256, 0, stream>>>(k_w, kw_bf, 128 * 128);
    cvt_f32_bf16<<<(128 * 128 + 255) / 256, 256, 0, stream>>>(proj_w, pw_bf, 128 * 128);
    cvt_f32_bf16<<<(512 * 128 + 255) / 256, 256, 0, stream>>>(fc1_w, f1w_bf, 512 * 128);
    cvt_f32_bf16<<<(128 * 512 + 255) / 256, 256, 0, stream>>>(fc2_w, f2w_bf, 128 * 512);

    // ---- norm1(x) ----
    ln_bf16<<<NP / 8, 256, 0, stream>>>(x, n1_w, n1_b, xn_bf);

    // ---- qv = xn @ qv_w^T + qv_b  (M=16384, N=256, K=128) ----
    gemm_wmma_bf16<128, GM_STORE_BF16><<<dim3(64, 4), 256, 0, stream>>>(
        xn_bf, qvw_bf, qv_b, nullptr, qvbf, nullptr, 256);

    // ---- k = y @ k_w^T + k_b  (N=128, K=128) ----
    gemm_wmma_bf16<128, GM_STORE_BF16><<<dim3(64, 2), 256, 0, stream>>>(
        y_bf, kw_bf, k_b, nullptr, k_bf, nullptr, 128);

    // ---- neighborhood attention ----
    natten49<<<dim3(16, 16, 4), 64, 0, stream>>>(qvbf, k_bf, rpb, ao_bf);

    // ---- x1 = x + attn_out @ proj_w^T + proj_b ----
    gemm_wmma_bf16<128, GM_RESID_F32><<<dim3(64, 2), 256, 0, stream>>>(
        ao_bf, pw_bf, proj_b, x, nullptr, x1, 128);

    // ---- norm2(x1) ----
    ln_bf16<<<NP / 8, 256, 0, stream>>>(x1, n2_w, n2_b, x1n_bf);

    // ---- h = gelu(x1n @ fc1_w^T + fc1_b)  (N=512, K=128) ----
    gemm_wmma_bf16<128, GM_GELU_BF16><<<dim3(64, 8), 256, 0, stream>>>(
        x1n_bf, f1w_bf, fc1_b, nullptr, h_bf, nullptr, 512);

    // ---- out = x1 + h @ fc2_w^T + fc2_b  (N=128, K=512) ----
    gemm_wmma_bf16<512, GM_RESID_F32><<<dim3(64, 2), 256, 0, stream>>>(
        h_bf, f2w_bf, fc2_b, x1, nullptr, (float*)d_out, 128);
}
